// IBertIntermediate_16106127360549
// MI455X (gfx1250) — 8-bit matrix operand order repaired and checked by reading, compiled, not run on hardware
//
#include <hip/hip_runtime.h>
#include <hip/hip_bf16.h>
#include <math.h>

#define GM 16384
#define GK 1024
#define GN 4096

typedef __attribute__((ext_vector_type(8))) int v8i;

__global__ void ibert_prep_weights(const float* __restrict__ w,
                                   const float* __restrict__ bias,
                                   const float* __restrict__ sf_in_p,
                                   signed char* __restrict__ w8,
                                   float* __restrict__ bintf,
                                   float* __restrict__ gbv,
                                   float* __restrict__ gcv,
                                   float* __restrict__ gsh,
                                   float* __restrict__ gsfout) {
    const int o   = blockIdx.x;
    const int tid = threadIdx.x;
    const float* row = w + (size_t)o * GK;

    float mx = 0.0f;
    for (int i = tid; i < GK; i += 256) mx = fmaxf(mx, fabsf(row[i]));

    __shared__ float red[256];
    red[tid] = mx;
    __syncthreads();
    for (int s = 128; s > 0; s >>= 1) {
        if (tid < s) red[tid] = fmaxf(red[tid], red[tid + s]);
        __syncthreads();
    }
    __shared__ float s_wsf;
    if (tid == 0) s_wsf = fmaxf(red[0], 1e-8f) * (1.0f / 127.0f);
    __syncthreads();
    const float wsf = s_wsf;
    const float inv_wsf = 1.0f / wsf;

    for (int i = tid; i < GK; i += 256) {
        float q = rintf(row[i] * inv_wsf);
        q = fminf(fmaxf(q, -128.0f), 127.0f);
        w8[(size_t)o * GK + i] = (signed char)(int)q;
    }

    if (tid == 0) {
        const float sf_in = sf_in_p[0];
        const float sf    = wsf * sf_in;
        bintf[o]  = rintf(bias[o] / sf);
        const float sfe   = sf * (1.0f / 1.4142f);
        gbv[o]    = floorf(-1.769f / sfe);
        gcv[o]    = floorf((1.0f / -0.2888f) / (sfe * sfe));
        const float sfsig = sfe * sfe * (-0.2888f) * 16384.0f;
        gsh[o]    = floorf(1.0f / sfsig);
        gsfout[o] = sf * sfsig * 0.5f;
    }
}

__global__ void ibert_quant_acts(const float* __restrict__ x,
                                 const float* __restrict__ sf_in_p,
                                 signed char* __restrict__ x8,
                                 int* __restrict__ maxbits) {
    if (blockIdx.x == 0 && threadIdx.x == 0) *maxbits = 0;
    const float inv_sf = 1.0f / sf_in_p[0];
    const size_t idx = ((size_t)blockIdx.x * 256 + threadIdx.x) * 4;
    const float4 v = *(const float4*)(x + idx);
    int q0 = (int)fminf(fmaxf(rintf(v.x * inv_sf), -128.0f), 127.0f);
    int q1 = (int)fminf(fmaxf(rintf(v.y * inv_sf), -128.0f), 127.0f);
    int q2 = (int)fminf(fmaxf(rintf(v.z * inv_sf), -128.0f), 127.0f);
    int q3 = (int)fminf(fmaxf(rintf(v.w * inv_sf), -128.0f), 127.0f);
    unsigned packed = (unsigned)(q0 & 0xFF) | ((unsigned)(q1 & 0xFF) << 8) |
                      ((unsigned)(q2 & 0xFF) << 16) | ((unsigned)(q3 & 0xFF) << 24);
    *(unsigned*)(x8 + idx) = packed;
}

__global__ void __launch_bounds__(256)
ibert_gemm_gelu_wmma(const signed char* __restrict__ x8,
                     const signed char* __restrict__ w8,
                     const float* __restrict__ bintf,
                     const float* __restrict__ gbv,
                     const float* __restrict__ gcv,
                     const float* __restrict__ gsh,
                     const float* __restrict__ gsfout,
                     float* __restrict__ out,
                     int* __restrict__ maxbits) {
    const int tid  = threadIdx.x;
    const int lane = tid & 31;
    const int wid  = tid >> 5;

    const int nblk = GN / 128;
    const int bx = blockIdx.x % nblk;
    const int by = blockIdx.x / nblk;

    const int m0 = by * 64  + (wid >> 2) * 32;
    const int n0 = bx * 128 + (wid & 3) * 32;

    const int ml  = lane & 15;
    const int kh  = (lane >> 4) * 16;
    const int kh2 = (lane >> 4) * 16;

    v8i acc[2][2] = {};

    for (int kc = 0; kc < GK; kc += 64) {
        v8i a[2], b[2];
#pragma unroll
        for (int ti = 0; ti < 2; ++ti) {
            const signed char* base =
                x8 + (size_t)(m0 + ti * 16 + ml) * GK + kc + kh;
            const uint2 p0 = *(const uint2*)(base + 0);
            const uint2 p1 = *(const uint2*)(base + 8);
            const uint2 p2 = *(const uint2*)(base + 32);
            const uint2 p3 = *(const uint2*)(base + 40);
            a[ti][0] = p0.x; a[ti][1] = p0.y;
            a[ti][2] = p1.x; a[ti][3] = p1.y;
            a[ti][4] = p2.x; a[ti][5] = p2.y;
            a[ti][6] = p3.x; a[ti][7] = p3.y;
        }
#pragma unroll
        for (int tj = 0; tj < 2; ++tj) {
            const signed char* base =
                w8 + (size_t)(n0 + tj * 16 + ml) * GK + kc + kh2;
            const uint4 q0 = *(const uint4*)(base + 0);
            const uint4 q1 = *(const uint4*)(base + 32);
            b[tj][0] = q0.x; b[tj][1] = q0.y; b[tj][2] = q0.z; b[tj][3] = q0.w;
            b[tj][4] = q1.x; b[tj][5] = q1.y; b[tj][6] = q1.z; b[tj][7] = q1.w;
        }
#pragma unroll
        for (int ti = 0; ti < 2; ++ti)
#pragma unroll
            for (int tj = 0; tj < 2; ++tj)
                acc[ti][tj] = __builtin_amdgcn_wmma_i32_16x16x64_iu8(
                     true, a[ti],  true, b[tj],
                    acc[ti][tj],  false,  false);
    }

    float lmax = 0.0f;
#pragma unroll
    for (int ti = 0; ti < 2; ++ti) {
        const int mrow = m0 + ti * 16 + (lane >> 4) * 8;
#pragma unroll
        for (int tj = 0; tj < 2; ++tj) {
            const int n = n0 + tj * 16 + ml;
            const float bi = bintf[n];
            const float gb = gbv[n];
            const float gc = gcv[n];
            const float sh = gsh[n];
            const float so = gsfout[n];
#pragma unroll
            for (int r = 0; r < 8; ++r) {
                const float xi = (float)acc[ti][tj][r] + bi;
                const float sgn = (xi > 0.0f) ? 1.0f : ((xi < 0.0f) ? -1.0f : 0.0f);
                const float ab = fminf(fabsf(xi), -gb);
                const float t  = ab + gb;
                float y = sgn * (t * t + gc);
                y = floorf(y * (1.0f / 16384.0f));
                const float x2 = xi * (y + sh);
                const float h2 = x2 * so;
                __builtin_nontemporal_store(h2, &out[(size_t)(mrow + r) * GN + n]);
                lmax = fmaxf(lmax, fabsf(h2));
            }
        }
    }

    __shared__ float red[256];
    red[tid] = lmax;
    __syncthreads();
    for (int s = 128; s > 0; s >>= 1) {
        if (tid < s) red[tid] = fmaxf(red[tid], red[tid + s]);
        __syncthreads();
    }
    if (tid == 0)
        atomicMax(maxbits, __float_as_int(red[0]));
}

__global__ void ibert_requant(float* __restrict__ out,
                              const float* __restrict__ gsfout,
                              const int* __restrict__ maxbits,
                              float* __restrict__ sf_slot) {
    const float maxabs = __int_as_float(*maxbits);
    const float sf2 = fmaxf(maxabs, 1e-8f) * (1.0f / 127.0f);
    const float inv_sf2 = 1.0f / sf2;

    const size_t idx = ((size_t)blockIdx.x * 256 + threadIdx.x) * 4;
#pragma unroll
    for (int j = 0; j < 4; ++j) {
        const size_t i = idx + j;
        const int n = (int)(i & (GN - 1));
        const float presf = gsfout[n];
        const float h = __builtin_nontemporal_load(&out[i]);
        const float z = rintf(h / presf);
        int e;
        const float m = frexpf(presf * inv_sf2, &e);
        const float mi = rintf(m * 2147483648.0f);
        float q = rintf(ldexpf(z * mi, e - 31));
        q = fminf(fmaxf(q, -128.0f), 127.0f);
        __builtin_nontemporal_store(q * sf2, &out[i]);
    }
    if (blockIdx.x == 0 && threadIdx.x == 0) sf_slot[0] = sf2;
}

extern "C" void kernel_launch(void* const* d_in, const int* in_sizes, int n_in,
                              void* d_out, int out_size, void* d_ws, size_t ws_size,
                              hipStream_t stream) {
    const float* x     = (const float*)d_in[0];
    const float* sf_in = (const float*)d_in[1];
    const float* w     = (const float*)d_in[2];
    const float* bias  = (const float*)d_in[3];
    float* out = (float*)d_out;

    unsigned char* ws = (unsigned char*)d_ws;
    signed char* w8 = (signed char*)ws;
    signed char* x8 = (signed char*)(ws + (size_t)GN * GK);
    float* bintf  = (float*)(ws + (size_t)GN * GK + (size_t)GM * GK);
    float* gbv    = bintf  + GN;
    float* gcv    = gbv    + GN;
    float* gsh    = gcv    + GN;
    float* gsfout = gsh    + GN;
    int*   maxbits = (int*)(gsfout + GN);
    float* sf_slot = out + (size_t)GM * GN;

    ibert_prep_weights<<<GN, 256, 0, stream>>>(w, bias, sf_in, w8,
                                               bintf, gbv, gcv, gsh, gsfout);
    ibert_quant_acts<<<(GM * GK) / (256 * 4), 256, 0, stream>>>(x, sf_in, x8, maxbits);
    ibert_gemm_gelu_wmma<<<(GM / 64) * (GN / 128), 256, 0, stream>>>(
        x8, w8, bintf, gbv, gcv, gsh, gsfout, out, maxbits);
    ibert_requant<<<((size_t)GM * GN) / (256 * 4), 256, 0, stream>>>(
        out, gsfout, maxbits, sf_slot);
}
